// MultiheadAttention_14319420964900
// MI455X (gfx1250) — matrix operand order repaired and checked by reading, not compiled, not run on hardware
//
#include <hip/hip_runtime.h>
#include <hip/hip_bf16.h>

// MI455X / gfx1250, wave32. Matmuls via v_wmma_f32_16x16x32_f16; flash
// attention K/V tiles staged LDS-side with global_load_async_to_lds_b128
// (ASYNCcnt) and double buffering. Softmax runs in log2 domain (exp2).

typedef __attribute__((ext_vector_type(16))) _Float16 v16h;
typedef __attribute__((ext_vector_type(8)))  _Float16 v8h;
typedef __attribute__((ext_vector_type(4)))  _Float16 v4h;
typedef __attribute__((ext_vector_type(8)))  float    v8f;

#define DEV static __device__ __forceinline__

DEV v8f wmma_f16(v16h a, v16h b, v8f c) {
  return __builtin_amdgcn_wmma_f32_16x16x32_f16(false, a, false, b, (short)0, c,
                                                false, false);
}

// A fragment (16x32 f16): lane l holds row (l&15); elems 0..15 = K c..c+15,
// c = (l>>4)*16.
DEV v16h load_a_frag(const _Float16* A, int lda, int row0, int k0) {
  const int lane = threadIdx.x & 31;
  const _Float16* p =
      A + (size_t)(row0 + (lane & 15)) * lda + k0 + ((lane >> 4) << 4);
  v8h lo = *(const v8h*)p;
  v8h hi = *(const v8h*)(p + 8);
  v16h r;
#pragma unroll
  for (int i = 0; i < 8; ++i) { r[i] = lo[i]; r[i + 8] = hi[i]; }
  return r;
}

// B fragment (32x16 f16), B[k][n] = Wt[n][k]: lane l holds col (l&15),
// K = k0 + (l>>4)*16 .. +15 -> one 32B load.
DEV v16h load_b_frag(const _Float16* Wt, int ldw, int col0, int k0) {
  const int lane = threadIdx.x & 31;
  return *(const v16h*)(Wt + (size_t)(col0 + (lane & 15)) * ldw + k0 +
                        ((lane >> 4) << 4));
}

DEV void wait_dscnt0() { asm volatile("s_wait_dscnt 0" ::: "memory"); }

// Fragment set for one 32x64 x k-step of GEMM (ping-pong, no reg copies).
struct Frags {
  v16h a0, a1, b[4];
};

DEV Frags load_frags(const _Float16* A, const _Float16* W, int K, int mrow,
                     int ncol0, int k0) {
  Frags f;
  f.a0 = load_a_frag(A, K, mrow, k0);
  f.a1 = load_a_frag(A, K, mrow + 16, k0);
#pragma unroll
  for (int t = 0; t < 4; ++t) f.b[t] = load_b_frag(W, K, ncol0 + 16 * t, k0);
  return f;
}

DEV void compute_frags(v8f acc[2][4], const Frags& f) {
#pragma unroll
  for (int t = 0; t < 4; ++t) acc[0][t] = wmma_f16(f.a0, f.b[t], acc[0][t]);
#pragma unroll
  for (int t = 0; t < 4; ++t) acc[1][t] = wmma_f16(f.a1, f.b[t], acc[1][t]);
}

// ---------------------------------------------------------------- converts
__global__ void cvt_f32_f16(const float* __restrict__ src,
                            _Float16* __restrict__ dst, int n) {
  int i = (blockIdx.x * blockDim.x + threadIdx.x) * 4;
  if (i < n) {
    float4 v = *(const float4*)(src + i);
    v4h o;
    o[0] = (_Float16)v.x; o[1] = (_Float16)v.y;
    o[2] = (_Float16)v.z; o[3] = (_Float16)v.w;
    *(v4h*)(dst + i) = o;
  }
}

// ------------------------------------------------------- fused QKV GEMM
// M=4096 (row = l*2 + b), N=3072, K=1024, ping-pong pipelined, 32x64/wave.
// Q epilogue scale folds softmax log2e: 0.125 * 1.4426950408889634.
__global__ __launch_bounds__(128) void qkv_gemm(
    const _Float16* __restrict__ q16, const _Float16* __restrict__ k16,
    const _Float16* __restrict__ v16m, const _Float16* __restrict__ w16,
    const float* __restrict__ bias, _Float16* __restrict__ qp,
    _Float16* __restrict__ kp, _Float16* __restrict__ vp) {
  const int K = 1024;
  const int ncol0 = blockIdx.x * 64;
  const int mrow = blockIdx.y * 128 + (threadIdx.x >> 5) * 32;
  const int sel = ncol0 >> 10;  // 0:q 1:k 2:v
  const _Float16* A = (sel == 0) ? q16 : ((sel == 1) ? k16 : v16m);

  v8f acc[2][4] = {};
  Frags X = load_frags(A, w16, K, mrow, ncol0, 0);
  // steady state: 2 k-steps per iteration, no fragment copies
  int k0 = 32;
  for (; k0 <= K - 64; k0 += 64) {
    __builtin_prefetch(A + (size_t)mrow * K + k0 + 224, 0, 3);
    Frags Y = load_frags(A, w16, K, mrow, ncol0, k0);
    compute_frags(acc, X);
    X = load_frags(A, w16, K, mrow, ncol0, k0 + 32);
    compute_frags(acc, Y);
  }
  compute_frags(acc, X);
  {
    Frags Y = load_frags(A, w16, K, mrow, ncol0, K - 32);
    compute_frags(acc, Y);
  }

  const int lane = threadIdx.x & 31;
#pragma unroll
  for (int mi = 0; mi < 2; ++mi) {
    const int rbase = mrow + mi * 16 + ((lane >> 4) << 3);
#pragma unroll
    for (int t = 0; t < 4; ++t) {
      int n = ncol0 + 16 * t + (lane & 15);
      int e = n & 1023;
      int g = e >> 6, dch = e & 63;
      float bv = bias[n];
#pragma unroll
      for (int i = 0; i < 8; ++i) {
        int m = rbase + i;
        int l = m >> 1, bb = m & 1;
        int bh = bb * 16 + g;
        float val = acc[mi][t][i] + bv;
        size_t idx = ((size_t)bh * 2048 + l) * 64 + dch;
        if (sel == 0)
          qp[idx] = (_Float16)(val * (0.125f * 1.44269504088896f));
        else if (sel == 1)
          kp[idx] = (_Float16)val;
        else
          vp[idx] = (_Float16)val;
      }
    }
  }
}

// --------------------------------------------------- flash dilated attention
// Head g at ratio DR attends over {p : p % DR == r}, r = g*DR/16.
// One wave / 16 query rows. K & V 32x64 tiles async-copied to LDS
// (double-buffered, ASYNCcnt), fragments sourced from LDS. Scores are in
// log2 domain (Q pre-scaled by log2e), so softmax uses exp2 directly.
template <int DR>
__global__ __launch_bounds__(32) void flash_attn(
    const _Float16* __restrict__ qp, const _Float16* __restrict__ kp,
    const _Float16* __restrict__ vp, float* __restrict__ oacc) {
  constexpr int L = 2048, D = 64;
  constexpr int LS = L / DR;
  constexpr int NT = LS / 32;
  const int bh = blockIdx.y;
  const int g = bh & 15;
  const int r = (g * DR) >> 4;
  const int q0 = blockIdx.x * 16;
  const int lane = threadIdx.x & 31;

  __shared__ _Float16 ktile[2][32][64];
  __shared__ _Float16 vtile[2][32][64];
  __shared__ _Float16 lds_p[16][32];

  const _Float16* Q  = qp + (size_t)bh * L * D;
  const _Float16* Km = kp + (size_t)bh * L * D;
  const _Float16* Vm = vp + (size_t)bh * L * D;

  const unsigned kbase = (unsigned)(size_t)&ktile[0][0][0];
  const unsigned vbase = (unsigned)(size_t)&vtile[0][0][0];

  // async-stage one 32x64 K tile + V tile: 256 16B chunks each / 32 lanes
  auto issue_tile = [&](int jt, int buf) {
    const int base_j = jt * 32;
#pragma unroll
    for (int it = 0; it < 8; ++it) {
      int c = it * 32 + lane;
      int jj = c >> 3, ch = c & 7;
      int p = (base_j + jj) * DR + r;
      unsigned long long ka =
          (unsigned long long)(Km + (size_t)p * D + ch * 8);
      unsigned long long va =
          (unsigned long long)(Vm + (size_t)p * D + ch * 8);
      unsigned off = (unsigned)(buf * 4096 + jj * 128 + ch * 16);
      unsigned ko = kbase + off, vo = vbase + off;
      asm volatile("global_load_async_to_lds_b128 %0, %1, off" ::"v"(ko),
                   "v"(ka)
                   : "memory");
      asm volatile("global_load_async_to_lds_b128 %0, %1, off" ::"v"(vo),
                   "v"(va)
                   : "memory");
    }
  };

  // Q A-fragments (d=0..31, 32..63)
  v16h aq0, aq1;
  {
    int p = (q0 + (lane & 15)) * DR + r;
    const _Float16* qr = Q + (size_t)p * D + ((lane >> 4) << 4);
    v8h x0 = *(const v8h*)qr;
    v8h x1 = *(const v8h*)(qr + 8);
    v8h x2 = *(const v8h*)(qr + 32);
    v8h x3 = *(const v8h*)(qr + 40);
#pragma unroll
    for (int i = 0; i < 8; ++i) {
      aq0[i] = x0[i]; aq0[i + 8] = x1[i];
      aq1[i] = x2[i]; aq1[i + 8] = x3[i];
    }
  }

  issue_tile(0, 0);

  v8f o[4] = {};
  v8f m_st, l_st;
#pragma unroll
  for (int i = 0; i < 8; ++i) { m_st[i] = -1e30f; l_st[i] = 0.f; }

  for (int jt = 0; jt < NT; ++jt) {
    const int cur = jt & 1;
    if (jt + 1 < NT) {
      wait_dscnt0();               // buf cur^1 reads (tile jt-1) retired
      issue_tile(jt + 1, cur ^ 1); // prefetch next tile while computing
      asm volatile("s_wait_asynccnt 0x10" ::: "memory");  // tile jt landed
    } else {
      asm volatile("s_wait_asynccnt 0x0" ::: "memory");
    }

    // ---- S = Q K^T from LDS K tile (contiguous 32B ds reads)
    v8f s0 = {}, s1 = {};
    {
      const _Float16* kr0 = &ktile[cur][lane & 15][(lane >> 4) << 4];
      const _Float16* kr1 = &ktile[cur][16 + (lane & 15)][(lane >> 4) << 4];
      v16h kb;
      kb = *(const v16h*)kr0;        s0 = wmma_f16(aq0, kb, s0);
      kb = *(const v16h*)(kr0 + 32); s0 = wmma_f16(aq1, kb, s0);
      kb = *(const v16h*)kr1;        s1 = wmma_f16(aq0, kb, s1);
      kb = *(const v16h*)(kr1 + 32); s1 = wmma_f16(aq1, kb, s1);
    }

    // ---- online softmax in log2 domain (rows per-VGPR, cols across lanes)
    v8f mx;
#pragma unroll
    for (int i = 0; i < 8; ++i) mx[i] = fmaxf(s0[i], s1[i]);
#pragma unroll
    for (int m = 1; m < 16; m <<= 1)
#pragma unroll
      for (int i = 0; i < 8; ++i)
        mx[i] = fmaxf(mx[i], __shfl_xor(mx[i], m, 32));

    v8f scl;
#pragma unroll
    for (int i = 0; i < 8; ++i) {
      float mn = fmaxf(m_st[i], mx[i]);
      scl[i] = exp2f(m_st[i] - mn);
      s0[i] = exp2f(s0[i] - mn);
      s1[i] = exp2f(s1[i] - mn);
      m_st[i] = mn;
    }
    v8f rs;
#pragma unroll
    for (int i = 0; i < 8; ++i) rs[i] = s0[i] + s1[i];
#pragma unroll
    for (int m = 1; m < 16; m <<= 1)
#pragma unroll
      for (int i = 0; i < 8; ++i) rs[i] += __shfl_xor(rs[i], m, 32);
#pragma unroll
    for (int i = 0; i < 8; ++i) {
      l_st[i] = l_st[i] * scl[i] + rs[i];
      o[0][i] *= scl[i]; o[1][i] *= scl[i];
      o[2][i] *= scl[i]; o[3][i] *= scl[i];
    }

    // ---- P: C-layout -> A-layout via LDS
    {
      int row = (lane >> 4) << 3, c = lane & 15;
#pragma unroll
      for (int i = 0; i < 8; ++i) {
        lds_p[row + i][c] = (_Float16)s0[i];
        lds_p[row + i][c + 16] = (_Float16)s1[i];
      }
    }
    wait_dscnt0();
    v16h pa;
    {
      int row = lane & 15, c = (lane >> 4) << 4;
#pragma unroll
      for (int i = 0; i < 8; ++i) {
        pa[i] = lds_p[row][c + i];
        pa[i + 8] = lds_p[row][c + 8 + i];
      }
    }

    // ---- O += P V from LDS V tile
#pragma unroll
    for (int t = 0; t < 4; ++t) {
      int dc = t * 16 + (lane & 15);
      int jb = (lane >> 4) << 4;
      v16h vb;
#pragma unroll
      for (int jj = 0; jj < 16; ++jj) vb[jj] = vtile[cur][jb + jj][dc];
      o[t] = wmma_f16(pa, vb, o[t]);
    }
  }

  // ---- scatter-accumulate to dense [bh][p][64]
  {
    int rbase = (lane >> 4) << 3, n = lane & 15;
#pragma unroll
    for (int i = 0; i < 8; ++i) {
      int p = (q0 + rbase + i) * DR + r;
      float inv = 1.f / l_st[i];
      float* orow = oacc + ((size_t)bh * L + p) * D;
      orow[n]      += o[0][i] * inv;
      orow[n + 16] += o[1][i] * inv;
      orow[n + 32] += o[2][i] * inv;
      orow[n + 48] += o[3][i] * inv;
    }
  }
}

// ---------------------------------------------- acc [bh][p][d] -> f16 rows
__global__ void acc_to_h(const float* __restrict__ acc,
                         _Float16* __restrict__ attn16) {
  int idx = blockIdx.x * blockDim.x + threadIdx.x;  // 4096*1024
  int m = idx >> 10, e = idx & 1023;
  int l = m >> 1, b = m & 1;
  int g = e >> 6, dch = e & 63;
  attn16[idx] =
      (_Float16)acc[(((size_t)(b * 16 + g)) * 2048 + l) * 64 + dch];
}

// ------------------------------------------------------- output projection
__global__ __launch_bounds__(128) void out_gemm(
    const _Float16* __restrict__ a16, const _Float16* __restrict__ ow16,
    const float* __restrict__ ob, float* __restrict__ out) {
  const int K = 1024;
  const int ncol0 = blockIdx.x * 64;
  const int mrow = blockIdx.y * 128 + (threadIdx.x >> 5) * 32;

  v8f acc[2][4] = {};
  Frags X = load_frags(a16, ow16, K, mrow, ncol0, 0);
  int k0 = 32;
  for (; k0 <= K - 64; k0 += 64) {
    __builtin_prefetch(a16 + (size_t)mrow * K + k0 + 224, 0, 3);
    Frags Y = load_frags(a16, ow16, K, mrow, ncol0, k0);
    compute_frags(acc, X);
    X = load_frags(a16, ow16, K, mrow, ncol0, k0 + 32);
    compute_frags(acc, Y);
  }
  compute_frags(acc, X);
  {
    Frags Y = load_frags(a16, ow16, K, mrow, ncol0, K - 32);
    compute_frags(acc, Y);
  }

  const int lane = threadIdx.x & 31;
#pragma unroll
  for (int mi = 0; mi < 2; ++mi) {
    const int rbase = mrow + mi * 16 + ((lane >> 4) << 3);
#pragma unroll
    for (int t = 0; t < 4; ++t) {
      int n = ncol0 + 16 * t + (lane & 15);
      float bv = ob[n];
#pragma unroll
      for (int i = 0; i < 8; ++i)
        out[(size_t)(rbase + i) * 1024 + n] = acc[mi][t][i] + bv;
    }
  }
}

extern "C" void kernel_launch(void* const* d_in, const int* in_sizes, int n_in,
                              void* d_out, int out_size, void* d_ws,
                              size_t ws_size, hipStream_t stream) {
  (void)in_sizes; (void)n_in; (void)out_size; (void)ws_size;
  const float* q  = (const float*)d_in[0];
  const float* k  = (const float*)d_in[1];
  const float* v  = (const float*)d_in[2];
  const float* w  = (const float*)d_in[3];   // (3072,1024)
  const float* bi = (const float*)d_in[4];   // (3072,)
  const float* ow = (const float*)d_in[5];   // (1024,1024)
  const float* ob = (const float*)d_in[6];   // (1024,)

  char* ws = (char*)d_ws;
  const size_t MB = 1ull << 20;
  _Float16* q16    = (_Float16*)(ws);
  _Float16* k16    = (_Float16*)(ws + 8 * MB);
  _Float16* v16b   = (_Float16*)(ws + 16 * MB);
  _Float16* w16    = (_Float16*)(ws + 24 * MB);
  _Float16* ow16   = (_Float16*)(ws + 30 * MB);
  _Float16* qp     = (_Float16*)(ws + 32 * MB);
  _Float16* kp     = (_Float16*)(ws + 40 * MB);
  _Float16* vp     = (_Float16*)(ws + 48 * MB);
  float*    acc    = (float*)   (ws + 56 * MB);
  _Float16* attn16 = (_Float16*)(ws + 72 * MB);

  const int NQ = 4096 * 1024;
  cvt_f32_f16<<<NQ / 1024, 256, 0, stream>>>(q, q16, NQ);
  cvt_f32_f16<<<NQ / 1024, 256, 0, stream>>>(k, k16, NQ);
  cvt_f32_f16<<<NQ / 1024, 256, 0, stream>>>(v, v16b, NQ);
  cvt_f32_f16<<<(3072 * 1024) / 1024, 256, 0, stream>>>(w, w16, 3072 * 1024);
  cvt_f32_f16<<<(1024 * 1024) / 1024, 256, 0, stream>>>(ow, ow16, 1024 * 1024);

  hipMemsetAsync(acc, 0, 32ull * 2048 * 64 * sizeof(float), stream);

  qkv_gemm<<<dim3(48, 32), 128, 0, stream>>>(q16, k16, v16b, w16, bi, qp, kp,
                                             vp);

  flash_attn<1><<<dim3(128, 32), 32, 0, stream>>>(qp, kp, vp, acc);
  flash_attn<2><<<dim3(64, 32), 32, 0, stream>>>(qp, kp, vp, acc);
  flash_attn<4><<<dim3(32, 32), 32, 0, stream>>>(qp, kp, vp, acc);
  flash_attn<8><<<dim3(16, 32), 32, 0, stream>>>(qp, kp, vp, acc);

  acc_to_h<<<(4096 * 1024) / 256, 256, 0, stream>>>(acc, attn16);

  out_gemm<<<dim3(16, 32), 128, 0, stream>>>(attn16, ow16, ob, (float*)d_out);
}
